// GraphSAGE_gc_32564442038491
// MI455X (gfx1250) — matrix operand order repaired and checked by reading, not compiled, not run on hardware
//
#include <hip/hip_runtime.h>
#include <hip/hip_bf16.h>

// ---------------- types ----------------
typedef _Float16 half_t;
typedef half_t v16h __attribute__((ext_vector_type(16)));
typedef half_t v8h  __attribute__((ext_vector_type(8)));
typedef float  v8f  __attribute__((ext_vector_type(8)));

#define MC 50000      // N_CELL
#define MG 512        // N_GENE
#define HF 256        // hidden
#define K0 512        // IN_FEATS
#define EC2G 400000
#define EG2C 400000
#define EG2G 40000

#define MODE_STORE 0
#define MODE_ADD   1
#define MODE_RELU  2

// ---------------- elementwise kernels (grid-stride, blockDim=256) ----------------
__global__ void k_cvt_f16(const float* __restrict__ in, half_t* __restrict__ out, size_t n) {
  for (size_t i = (size_t)blockIdx.x * 256 + threadIdx.x; i < n; i += (size_t)gridDim.x * 256)
    out[i] = (half_t)in[i];
}

__global__ void k_gather_embed(const float* __restrict__ embed, const int* __restrict__ ids,
                               float* __restrict__ xg, size_t n) { // n = MG*K0
  for (size_t i = (size_t)blockIdx.x * 256 + threadIdx.x; i < n; i += (size_t)gridDim.x * 256) {
    size_t r = i >> 9;            // /512
    size_t c = i & 511;
    xg[i] = embed[(size_t)ids[r] * K0 + c];
  }
}

__global__ void k_deg(const int* __restrict__ dst, float* __restrict__ deg, int E) {
  for (size_t i = (size_t)blockIdx.x * 256 + threadIdx.x; i < (size_t)E; i += (size_t)gridDim.x * 256)
    atomicAdd(&deg[dst[i]], 1.0f);
}

__global__ void k_meandiv(float* __restrict__ acc, const float* __restrict__ deg, size_t n) {
  for (size_t i = (size_t)blockIdx.x * 256 + threadIdx.x; i < n; i += (size_t)gridDim.x * 256)
    acc[i] /= fmaxf(deg[i >> 8], 1.0f);
}

__global__ void k_gcn_combine(const float* __restrict__ acc, const float* __restrict__ h,
                              const float* __restrict__ deg, float* __restrict__ out, size_t n) {
  for (size_t i = (size_t)blockIdx.x * 256 + threadIdx.x; i < n; i += (size_t)gridDim.x * 256)
    out[i] = (acc[i] + h[i]) / (deg[i >> 8] + 1.0f);
}

__global__ void k_add(float* __restrict__ a, const float* __restrict__ b, size_t n) {
  for (size_t i = (size_t)blockIdx.x * 256 + threadIdx.x; i < n; i += (size_t)gridDim.x * 256)
    a[i] += b[i];
}

__global__ void k_add_relu(float* __restrict__ a, const float* __restrict__ b, size_t n) {
  for (size_t i = (size_t)blockIdx.x * 256 + threadIdx.x; i < n; i += (size_t)gridDim.x * 256)
    a[i] += fmaxf(b[i], 0.0f);
}

__global__ void k_relu_copy(float* __restrict__ d, const float* __restrict__ s, size_t n) {
  for (size_t i = (size_t)blockIdx.x * 256 + threadIdx.x; i < n; i += (size_t)gridDim.x * 256)
    d[i] = fmaxf(s[i], 0.0f);
}

// ---------------- graph scatters: one block per edge, 256 lanes = 256 features ----------------
__global__ void k_scatter_wsum(const float* __restrict__ feat, const int* __restrict__ src,
                               const int* __restrict__ dst, const float* __restrict__ w,
                               float* __restrict__ acc) {
  int e = blockIdx.x;
  int f = threadIdx.x;
  int s = src[e], d = dst[e];
  float v = w[e] * feat[((size_t)s << 8) + f];
  atomicAdd(&acc[((size_t)d << 8) + f], v);
}

// values are w*relu(...) >= 0, acc pre-zeroed -> int bit compare == float compare
__global__ void k_scatter_wmax(const float* __restrict__ feat, const int* __restrict__ src,
                               const int* __restrict__ dst, const float* __restrict__ w,
                               float* __restrict__ acc) {
  int e = blockIdx.x;
  int f = threadIdx.x;
  int s = src[e], d = dst[e];
  float v = w[e] * feat[((size_t)s << 8) + f];
  atomicMax((int*)&acc[((size_t)d << 8) + f], __float_as_int(v));
}

// ---------------- WMMA GEMM: out[M,256] (op)= A16[M,K] @ W16[256,K]^T + bias ----------------
// 8 waves/block; each wave: one 16-row strip x full N=256 (16 accumulators).
__global__ __launch_bounds__(256)
void k_wmma_gemm256(const half_t* __restrict__ A, const half_t* __restrict__ W,
                    const float* __restrict__ bias, float* __restrict__ out,
                    int M, int K, int mode) {
  int wave = threadIdx.x >> 5;
  int lane = threadIdx.x & 31;
  int rowBase = (blockIdx.x * 8 + wave) * 16;
  if (rowBase >= M) return;               // wave-uniform exit (EXEC stays all-ones)
  int r  = lane & 15;
  int hh = lane >> 4;

  v8f acc[16];
#pragma unroll
  for (int t = 0; t < 16; ++t)
#pragma unroll
    for (int j = 0; j < 8; ++j) acc[t][j] = 0.0f;

  const half_t* arow = A + (size_t)(rowBase + r) * K;
  for (int k0 = 0; k0 < K; k0 += 32) {
    // A fragment: lane half hh -> K chunks {hh*8..hh*8+7, 16+hh*8..16+hh*8+7}
    v16h a;
    {
      const v8h* pa = (const v8h*)(arow + k0 + hh * 8);
      v8h lo = pa[0];
      v8h hi = pa[2];
#pragma unroll
      for (int i = 0; i < 8; ++i) { a[i] = lo[i]; a[8 + i] = hi[i]; }
    }
#pragma unroll
    for (int t = 0; t < 16; ++t) {
      // B fragment: column n = t*16+r, K chunks {hh*8..hh*8+7, 16+hh*8..16+hh*8+7}
      const v8h* pb = (const v8h*)(W + (size_t)(t * 16 + r) * K + k0 + hh * 8);
      v8h lo = pb[0];
      v8h hi = pb[2];
      v16h b;
#pragma unroll
      for (int i = 0; i < 8; ++i) { b[i] = lo[i]; b[8 + i] = hi[i]; }
      acc[t] = __builtin_amdgcn_wmma_f32_16x16x32_f16(
          false, a, false, b, (short)0, acc[t], false, false);
    }
  }

#pragma unroll
  for (int t = 0; t < 16; ++t) {
    int n = t * 16 + r;
    float bv = bias ? bias[n] : 0.0f;
#pragma unroll
    for (int j = 0; j < 8; ++j) {
      int row = rowBase + j + 8 * hh;
      size_t o = (size_t)row * 256 + n;
      float v = acc[t][j] + bv;
      if (mode == MODE_STORE)      out[o] = v;
      else if (mode == MODE_ADD)   out[o] += v;
      else                         out[o] = fmaxf(v, 0.0f);
    }
  }
}

// ---------------- host ----------------
extern "C" void kernel_launch(void* const* d_in, const int* in_sizes, int n_in,
                              void* d_out, int out_size, void* d_ws, size_t ws_size,
                              hipStream_t stream) {
  // ---- inputs (setup_inputs insertion order) ----
  const float* xc      = (const float*)d_in[0];
  const int*   geneids = (const int*)d_in[1];
  const int *c2g_s = (const int*)d_in[2], *c2g_d = (const int*)d_in[3];
  const int *g2c_s = (const int*)d_in[4], *g2c_d = (const int*)d_in[5];
  const int *g2g_s = (const int*)d_in[6], *g2g_d = (const int*)d_in[7];
  const float *w_c2g = (const float*)d_in[8], *w_g2c = (const float*)d_in[9],
              *w_g2g = (const float*)d_in[10];
  // params: jax pytree leaves, sorted dict keys:
  // conv1[c2g:Wn,Ws,b][g2c][g2g]=11..19 ; conv2[c2g:Wn,b][g2c][g2g]=20..25
  // conv3[c2g:Wn,Wp,Ws,b,bp][g2c][g2g]=26..40 ; conv4=41..49 ; conv5=50..55 ; conv6=56..70
  // embed=71 ; lin1[cell:W,b][gene:W,b]=72..75 ; lin2=76..79
  auto P = [&](int i) -> const float* { return (const float*)d_in[i]; };

  const size_t MCH = (size_t)MC * HF;     // 12.8M
  const size_t MGH = (size_t)MG * HF;     // 131072

  // ---- workspace carve ----
  char* wp = (char*)d_ws;
  auto alloc = [&](size_t bytes) -> void* {
    void* r = (void*)wp;
    wp += (bytes + 255) & ~(size_t)255;
    return r;
  };
  float* hc    = (float*)alloc(MCH * 4);
  float* tc    = (float*)alloc(MCH * 4);
  float* acc_c = (float*)alloc(MCH * 4);
  float* hp_c  = (float*)alloc(MCH * 4);
  float* hg    = (float*)alloc(MGH * 4);
  float* tg    = (float*)alloc(MGH * 4);
  float* acc_g = (float*)alloc(MGH * 4);
  float* hp_g  = (float*)alloc(MGH * 4);
  float* xg    = (float*)alloc((size_t)MG * K0 * 4);
  float* deg_cg = (float*)alloc(MG * 4);   // c2g dst = gene
  float* deg_gc = (float*)alloc((size_t)MC * 4); // g2c dst = cell
  float* deg_gg = (float*)alloc(MG * 4);   // g2g dst = gene
  half_t* a16 = (half_t*)alloc((size_t)MC * K0 * 2);
  half_t* g16 = (half_t*)alloc((size_t)MG * K0 * 2);
  half_t* w16 = (half_t*)alloc((size_t)HF * K0 * 2);

  // ---- launch helpers ----
  auto gs = [](size_t n) -> unsigned {
    size_t b = (n + 255) >> 8;
    return (unsigned)(b < 262144 ? b : 262144);
  };
  auto zero = [&](void* p, size_t bytes) { hipMemsetAsync(p, 0, bytes, stream); };
  auto cvt = [&](const float* s, half_t* d, size_t n) {
    k_cvt_f16<<<gs(n), 256, 0, stream>>>(s, d, n);
  };
  auto gemm = [&](const half_t* A, const float* Wf, const float* bias, float* out,
                  int M, int K, int mode) {
    cvt(Wf, w16, (size_t)HF * K);
    dim3 grid((M / 16 + 7) / 8);
    k_wmma_gemm256<<<grid, 256, 0, stream>>>(A, w16, bias, out, M, K, mode);
  };
  auto scat_sum = [&](const float* f, const int* s, const int* d, const float* w,
                      float* acc, int E) {
    k_scatter_wsum<<<E, 256, 0, stream>>>(f, s, d, w, acc);
  };
  auto scat_max = [&](const float* f, const int* s, const int* d, const float* w,
                      float* acc, int E) {
    k_scatter_wmax<<<E, 256, 0, stream>>>(f, s, d, w, acc);
  };
  auto meandiv = [&](float* acc, const float* deg, size_t n) {
    k_meandiv<<<gs(n), 256, 0, stream>>>(acc, deg, n);
  };
  auto gcnc = [&](const float* acc, const float* h, const float* deg, float* o, size_t n) {
    k_gcn_combine<<<gs(n), 256, 0, stream>>>(acc, h, deg, o, n);
  };
  auto addip = [&](float* a, const float* b, size_t n) {
    k_add<<<gs(n), 256, 0, stream>>>(a, b, n);
  };
  auto add_relu = [&](float* a, const float* b, size_t n) {
    k_add_relu<<<gs(n), 256, 0, stream>>>(a, b, n);
  };
  auto relu_copy = [&](float* d, const float* s, size_t n) {
    k_relu_copy<<<gs(n), 256, 0, stream>>>(d, s, n);
  };

  // ---- degrees (constant within a launch) ----
  zero(deg_cg, MG * 4); zero(deg_gc, (size_t)MC * 4); zero(deg_gg, MG * 4);
  k_deg<<<gs(EC2G), 256, 0, stream>>>(c2g_d, deg_cg, EC2G);
  k_deg<<<gs(EG2C), 256, 0, stream>>>(g2c_d, deg_gc, EG2C);
  k_deg<<<gs(EG2G), 256, 0, stream>>>(g2g_d, deg_gg, EG2G);

  // ---- gene embeddings + f16 inputs ----
  k_gather_embed<<<gs((size_t)MG * K0), 256, 0, stream>>>(P(71), geneids, xg, (size_t)MG * K0);
  cvt(xc, a16, (size_t)MC * K0);     // a16 = f16(x_cell) [MC,512]
  cvt(xg, g16, (size_t)MG * K0);     // g16 = f16(x_gene) [MG,512]

  // ================= conv1: SAGE 'mean', lin_before=True, 512->256, then relu =================
  // gene <- c2g
  gemm(a16, P(11), nullptr, hp_c, MC, K0, MODE_STORE);                 // h_src = xc @ Wn^T
  zero(acc_g, MGH * 4);
  scat_sum(hp_c, c2g_s, c2g_d, w_c2g, acc_g, EC2G);
  meandiv(acc_g, deg_cg, MGH);
  gemm(g16, P(12), P(13), tg, MG, K0, MODE_STORE);                     // xg @ Ws^T + b
  addip(tg, acc_g, MGH);
  // gene <- g2g
  gemm(g16, P(17), nullptr, hp_g, MG, K0, MODE_STORE);
  zero(acc_g, MGH * 4);
  scat_sum(hp_g, g2g_s, g2g_d, w_g2g, acc_g, EG2G);
  meandiv(acc_g, deg_gg, MGH);
  gemm(g16, P(18), P(19), tg, MG, K0, MODE_ADD);
  addip(tg, acc_g, MGH);
  // cell <- g2c
  gemm(g16, P(14), nullptr, hp_g, MG, K0, MODE_STORE);                 // h_src = xg @ Wn^T
  zero(acc_c, MCH * 4);
  scat_sum(hp_g, g2c_s, g2c_d, w_g2c, acc_c, EG2C);
  meandiv(acc_c, deg_gc, MCH);
  gemm(a16, P(15), P(16), tc, MC, K0, MODE_STORE);                     // xc @ Ws^T + b
  addip(tc, acc_c, MCH);
  relu_copy(hc, tc, MCH);
  relu_copy(hg, tg, MGH);

  // ================= generic layer helpers (256->256) =================
  auto conv_gcn = [&](int base) {  // per-edge: [Wn,b]; c2g=+0, g2c=+2, g2g=+4
    zero(acc_g, MGH * 4);
    scat_sum(hc, c2g_s, c2g_d, w_c2g, acc_g, EC2G);
    gcnc(acc_g, hg, deg_cg, hp_g, MGH);
    cvt(hp_g, g16, MGH);
    gemm(g16, P(base + 0), P(base + 1), tg, MG, HF, MODE_STORE);
    zero(acc_g, MGH * 4);
    scat_sum(hg, g2g_s, g2g_d, w_g2g, acc_g, EG2G);
    gcnc(acc_g, hg, deg_gg, hp_g, MGH);
    cvt(hp_g, g16, MGH);
    gemm(g16, P(base + 4), P(base + 5), tg, MG, HF, MODE_ADD);
    zero(acc_c, MCH * 4);
    scat_sum(hg, g2c_s, g2c_d, w_g2c, acc_c, EG2C);
    gcnc(acc_c, hc, deg_gc, hp_c, MCH);
    cvt(hp_c, a16, MCH);
    gemm(a16, P(base + 2), P(base + 3), tc, MC, HF, MODE_STORE);
    add_relu(hc, tc, MCH);
    add_relu(hg, tg, MGH);
  };

  auto conv_pool = [&](int base) { // per-edge: [Wn,Wp,Ws,b,bp]; c2g=+0, g2c=+5, g2g=+10
    // gene <- c2g
    cvt(hc, a16, MCH);                                            // a16 = f16(hc)
    gemm(a16, P(base + 1), P(base + 4), hp_c, MC, HF, MODE_RELU); // relu(hc@Wp^T+bp)
    zero(acc_g, MGH * 4);
    scat_max(hp_c, c2g_s, c2g_d, w_c2g, acc_g, EC2G);
    cvt(hg, g16, MGH);
    gemm(g16, P(base + 2), P(base + 3), tg, MG, HF, MODE_STORE);  // hg@Ws^T+b
    cvt(acc_g, g16, MGH);
    gemm(g16, P(base + 0), nullptr, tg, MG, HF, MODE_ADD);        // + neigh@Wn^T
    // gene <- g2g
    cvt(hg, g16, MGH);
    gemm(g16, P(base + 11), P(base + 14), hp_g, MG, HF, MODE_RELU);
    zero(acc_g, MGH * 4);
    scat_max(hp_g, g2g_s, g2g_d, w_g2g, acc_g, EG2G);
    gemm(g16, P(base + 12), P(base + 13), tg, MG, HF, MODE_ADD);  // g16 still f16(hg)
    cvt(acc_g, g16, MGH);
    gemm(g16, P(base + 10), nullptr, tg, MG, HF, MODE_ADD);
    // cell <- g2c
    cvt(hg, g16, MGH);
    gemm(g16, P(base + 6), P(base + 9), hp_g, MG, HF, MODE_RELU);
    zero(acc_c, MCH * 4);
    scat_max(hp_g, g2c_s, g2c_d, w_g2c, acc_c, EG2C);
    gemm(a16, P(base + 7), P(base + 8), tc, MC, HF, MODE_STORE);  // a16 still f16(hc)
    cvt(acc_c, a16, MCH);
    gemm(a16, P(base + 5), nullptr, tc, MC, HF, MODE_ADD);
    add_relu(hc, tc, MCH);
    add_relu(hg, tg, MGH);
  };

  auto conv_mean_nl = [&](int base) { // per-edge: [Wn,Ws,b]; c2g=+0, g2c=+3, g2g=+6
    zero(acc_g, MGH * 4);
    scat_sum(hc, c2g_s, c2g_d, w_c2g, acc_g, EC2G);
    meandiv(acc_g, deg_cg, MGH);
    cvt(acc_g, g16, MGH);
    gemm(g16, P(base + 0), nullptr, tg, MG, HF, MODE_STORE);      // neigh@Wn^T
    cvt(hg, g16, MGH);
    gemm(g16, P(base + 1), P(base + 2), tg, MG, HF, MODE_ADD);    // + hg@Ws^T + b
    zero(acc_g, MGH * 4);
    scat_sum(hg, g2g_s, g2g_d, w_g2g, acc_g, EG2G);
    meandiv(acc_g, deg_gg, MGH);
    cvt(acc_g, g16, MGH);
    gemm(g16, P(base + 6), nullptr, tg, MG, HF, MODE_ADD);
    cvt(hg, g16, MGH);
    gemm(g16, P(base + 7), P(base + 8), tg, MG, HF, MODE_ADD);
    zero(acc_c, MCH * 4);
    scat_sum(hg, g2c_s, g2c_d, w_g2c, acc_c, EG2C);
    meandiv(acc_c, deg_gc, MCH);
    cvt(acc_c, a16, MCH);
    gemm(a16, P(base + 3), nullptr, tc, MC, HF, MODE_STORE);
    cvt(hc, a16, MCH);
    gemm(a16, P(base + 4), P(base + 5), tc, MC, HF, MODE_ADD);
    add_relu(hc, tc, MCH);
    add_relu(hg, tg, MGH);
  };

  auto lin = [&](int wc, int bc, int wg, int bg, float* oc, float* og) {
    cvt(hc, a16, MCH);
    gemm(a16, P(wc), P(bc), oc, MC, HF, MODE_STORE);
    cvt(hg, g16, MGH);
    gemm(g16, P(wg), P(bg), og, MG, HF, MODE_STORE);
  };

  // ================= layers 2..6 + linears =================
  conv_gcn(20);            // conv2
  conv_pool(26);           // conv3
  lin(72, 73, 74, 75, hc, hg);   // lin1 (in-place; GEMM reads f16 copies)
  conv_mean_nl(41);        // conv4
  conv_gcn(50);            // conv5
  conv_pool(56);           // conv6

  float* out_c = (float*)d_out;
  float* out_g = out_c + MCH;
  lin(76, 77, 78, 79, out_c, out_g);  // lin2 -> outputs (cell, gene)
}
